// HybridSelfAttention_65481071396578
// MI455X (gfx1250) — compile-verified, matrix operand order checked by reading, not run on hardware
//
#include <hip/hip_runtime.h>

// ---------------------------------------------------------------------------
// HybridSelfAttention for MI455X (gfx1250, wave32, WMMA bf16 16x16x32)
// B=4, S=1024, E=512, H=16
//
// Round-4: same as round-3 (sched_group_barrier load/WMMA pipelines) with the
// non-constant group-size ternary split into literal-constant calls.
// ---------------------------------------------------------------------------

typedef __attribute__((ext_vector_type(16))) __bf16 bf16x16;
typedef __attribute__((ext_vector_type(8)))  float  v8f;
typedef unsigned short u16;
typedef unsigned int   u32;

#define B_ 4
#define S_ 1024
#define E_ 512
#define H_ 16

#define SGB  __builtin_amdgcn_sched_group_barrier
#define MASK_VMEM_READ 0x020
#define MASK_WMMA      0x008

// fp32 -> bf16, round-to-nearest-even
__device__ __forceinline__ u16 f2bf(float f) {
  u32 u = __builtin_bit_cast(u32, f);
  u32 r = u + 0x7FFFu + ((u >> 16) & 1u);
  return (u16)(r >> 16);
}

// Load one 16x32 bf16 WMMA operand fragment (A, or B from an [n][k] row-major
// source). CDNA5 16-bit A layout: lane L (L<16) holds row M=L,
// K in [0,8)∪[16,24); lane L+16 holds row M=L, K in [8,16)∪[24,32).
// => per lane: two contiguous 16-byte chunks (GLOBAL_LOAD_B128 x2).
__device__ __forceinline__ bf16x16 load_frag(const u16* __restrict__ base,
                                             int stride, int lane) {
  int r = lane & 15, hf = lane >> 4;
  const u16* p = base + (size_t)r * stride + hf * 8;
  union { uint4 q[2]; bf16x16 v; } f;
  f.q[0] = *(const uint4*)(p);        // K = hf*8 .. hf*8+7
  f.q[1] = *(const uint4*)(p + 16);   // K = hf*8+16 .. hf*8+23
  return f.v;
}

// tanh(x) = 1 - 2/(exp(2x)+1), via v_exp_f32 (exp2) + v_rcp_f32.
__device__ __forceinline__ float fast_tanh(float x) {
  float e = __builtin_amdgcn_exp2f(x * 2.885390081777927f); // 2*log2(e)
  return 1.0f - 2.0f * __builtin_amdgcn_rcpf(e + 1.0f);
}

// Pipeline shape for a 16-step WMMA chain fed by 2-load fragments:
// 6 loads (3 frags) ahead, then (1 WMMA, 2 loads) x13, then 3 WMMAs.
__device__ __forceinline__ void pipeline_16(void) {
  SGB(MASK_VMEM_READ, 6, 0);
#pragma unroll
  for (int i = 0; i < 13; ++i) {
    SGB(MASK_WMMA, 1, 0);
    SGB(MASK_VMEM_READ, 2, 0);
  }
  SGB(MASK_WMMA, 3, 0);
}

// ---------------------------------------------------------------------------
// fp32 -> bf16 elementwise convert
// ---------------------------------------------------------------------------
__global__ __launch_bounds__(256) void cvt_bf16(const float* __restrict__ in,
                                                u16* __restrict__ out, int n) {
  int i = blockIdx.x * 256 + threadIdx.x;
  if (i < n) out[i] = f2bf(in[i]);
}

// split W1 (H x 2E) into W1q (H x E) and W1k (H x E), bf16
__global__ __launch_bounds__(256) void split_w1(const float* __restrict__ W1,
                                                u16* __restrict__ W1q,
                                                u16* __restrict__ W1k) {
  int i = blockIdx.x * 256 + threadIdx.x;  // H_*E_ = 8192 elems
  if (i < H_ * E_) {
    int h = i / E_, e = i % E_;
    W1q[i] = f2bf(W1[h * 2 * E_ + e]);
    W1k[i] = f2bf(W1[h * 2 * E_ + E_ + e]);
  }
}

// ---------------------------------------------------------------------------
// QKV projection: [B*S x E] x [E x E]^T for Wq/Wk/Wv.
// One block per (b, 16-row s-tile); x-tile A fragments hoisted (16 frags,
// 128 VGPRs); 8 waves cover 3 matrices x 32 o-tiles = 96 output tiles.
// B fragments preloaded as an array; sched_group_barrier pipeline keeps
// 3 fragments (6 B128 loads) in flight per WMMA.
// ---------------------------------------------------------------------------
__global__ __launch_bounds__(256) void qkv_kernel(
    const u16* __restrict__ xbf,
    const u16* __restrict__ Wqb, const u16* __restrict__ Wkb,
    const u16* __restrict__ Wvb,
    u16* __restrict__ Qb, u16* __restrict__ Kb, u16* __restrict__ Vt) {
  int lane = threadIdx.x & 31, wave = threadIdx.x >> 5;
  int b  = blockIdx.x >> 6;
  int s0 = (blockIdx.x & 63) << 4;
  size_t rowbase = (size_t)(b * S_ + s0) * E_;

  bf16x16 a[16];
#pragma unroll
  for (int kk = 0; kk < 16; ++kk)
    a[kk] = load_frag(xbf + rowbase + kk * 32, E_, lane);

  int n = lane & 15, hf = lane >> 4;
  for (int t = wave; t < 96; t += 8) {
    int mat = t >> 5;            // 0=Q 1=K 2=V
    int o0  = (t & 31) << 4;
    const u16* W = (mat == 0) ? Wqb : (mat == 1 ? Wkb : Wvb);
    const u16* Wb = W + (size_t)o0 * E_;
    v8f c = {};
    bf16x16 bb[16];
#pragma unroll
    for (int kk = 0; kk < 16; ++kk)
      bb[kk] = load_frag(Wb + kk * 32, E_, lane);
#pragma unroll
    for (int kk = 0; kk < 16; ++kk)
      c = __builtin_amdgcn_wmma_f32_16x16x32_bf16(false, a[kk], false, bb[kk],
                                                  (short)0, c, false, false);
    pipeline_16();
    if (mat < 2) {
      u16* dst = (mat == 0) ? Qb : Kb;
#pragma unroll
      for (int r = 0; r < 8; ++r) {
        int m = r + 8 * hf;                       // C layout: VGPR r, half hf
        dst[rowbase + (size_t)m * E_ + o0 + n] = f2bf(c[r]);
      }
    } else {
#pragma unroll
      for (int r = 0; r < 8; ++r) {
        int m = r + 8 * hf;
        Vt[(size_t)(b * E_ + o0 + n) * S_ + s0 + m] = f2bf(c[r]);
      }
    }
  }
}

// ---------------------------------------------------------------------------
// hq = Q @ W1q^T (and hk = K @ W1k^T): [4096x512]x[512x16]. N=16 == one WMMA
// tile wide. Both streams preloaded; pipeline keeps 2 fragment pairs ahead.
// ---------------------------------------------------------------------------
__global__ __launch_bounds__(256) void hqk_kernel(
    const u16* __restrict__ src, const u16* __restrict__ W1b,
    float* __restrict__ hout) {
  int lane = threadIdx.x & 31, wave = threadIdx.x >> 5;
  int st = blockIdx.x * 8 + wave;            // 0..255 row tiles of 16
  size_t rowbase = (size_t)st * 16 * E_;
  v8f c = {};
  bf16x16 aa[16], bb[16];
#pragma unroll
  for (int kk = 0; kk < 16; ++kk) {
    aa[kk] = load_frag(src + rowbase + kk * 32, E_, lane);
    bb[kk] = load_frag(W1b + kk * 32, E_, lane);
  }
#pragma unroll
  for (int kk = 0; kk < 16; ++kk)
    c = __builtin_amdgcn_wmma_f32_16x16x32_bf16(false, aa[kk], false, bb[kk],
                                                (short)0, c, false, false);
  // 8 loads (2 a+b pairs) ahead, then (1 WMMA, 4 loads) x14, then 2 WMMAs.
  SGB(MASK_VMEM_READ, 8, 0);
#pragma unroll
  for (int i = 0; i < 14; ++i) {
    SGB(MASK_WMMA, 1, 0);
    SGB(MASK_VMEM_READ, 4, 0);
  }
  SGB(MASK_WMMA, 2, 0);

  int n = lane & 15, hf = lane >> 4;
#pragma unroll
  for (int r = 0; r < 8; ++r)
    hout[(size_t)(st * 16 + r + 8 * hf) * H_ + n] = c[r];
}

// ---------------------------------------------------------------------------
// Fused scores + est(tanh MLP) + full-row softmax -> attn (bf16).
// One block per (b, 16-row q-tile). 8 waves; Q fragments (full K=512) hoisted
// in registers; each wave handles 8 k-tiles of 16. Scores land in 64KB LDS;
// full-row softmax; bf16 attn out (L2-resident: 8MB << 192MB L2).
// sched_barrier(0) isolates the WMMA pipeline from the tanh/est VALU block.
// ---------------------------------------------------------------------------
__global__ __launch_bounds__(256) void attn_scores_kernel(
    const u16* __restrict__ Qb, const u16* __restrict__ Kb,
    const float* __restrict__ hq, const float* __restrict__ hk,
    const float* __restrict__ b1, const float* __restrict__ W2,
    const float* __restrict__ b2, u16* __restrict__ attn) {
  __shared__ float sS[16 * S_];   // 64 KB score tile
  __shared__ float sAux[256];     // hq tile, later reduction scratch
  __shared__ float sRow[32];      // rowmax[16], rowsum[16]

  int tid = threadIdx.x;
  int lane = tid & 31, wave = tid >> 5;
  int b  = blockIdx.x >> 6;
  int q0 = (blockIdx.x & 63) << 4;
  size_t qrow = (size_t)(b * S_ + q0);

  // stage hq[q0..q0+15][0..15] into LDS
  sAux[tid] = hq[(qrow + (tid >> 4)) * H_ + (tid & 15)];
  __syncthreads();

  bf16x16 a[16];
#pragma unroll
  for (int kk = 0; kk < 16; ++kk)
    a[kk] = load_frag(Qb + qrow * E_ + kk * 32, E_, lane);

  float w2r[16], b1r[16];
#pragma unroll
  for (int h = 0; h < 16; ++h) { w2r[h] = W2[h]; b1r[h] = b1[h]; }
  float b2v = b2[0];
  const float scale = 0.044194173824159216f;  // 1/sqrt(512)
  int n = lane & 15, hf = lane >> 4;

  for (int kt = wave; kt < 64; kt += 8) {
    int k0 = kt << 4;
    size_t krow = (size_t)(b * S_ + k0);
    const u16* Kbase = Kb + krow * E_;
    v8f c = {};
    bf16x16 bb[16];
#pragma unroll
    for (int kk = 0; kk < 16; ++kk)
      bb[kk] = load_frag(Kbase + kk * 32, E_, lane);
#pragma unroll
    for (int kk = 0; kk < 16; ++kk)
      c = __builtin_amdgcn_wmma_f32_16x16x32_bf16(false, a[kk], false, bb[kk],
                                                  (short)0, c, false, false);
    pipeline_16();
    __builtin_amdgcn_sched_barrier(0);   // keep est code out of the pipeline

    // est for this 16x16 tile: column n fixed per lane -> hoist hk row
    float hkr[16];
#pragma unroll
    for (int h = 0; h < 16; ++h) hkr[h] = hk[(krow + n) * H_ + h];
#pragma unroll
    for (int r = 0; r < 8; ++r) {
      int m = r + 8 * hf;
      float est = b2v;
#pragma unroll
      for (int h = 0; h < 16; ++h)
        est += w2r[h] * fast_tanh(sAux[m * 16 + h] + hkr[h] + b1r[h]);
      sS[m * S_ + k0 + n] = c[r] * scale + est;
    }
  }
  __syncthreads();

  // Row softmax over 1024 cols: thread (row, j) owns cols j, j+16, ...
  int row = tid >> 4, j = tid & 15;
  float mx = -3.4e38f;
#pragma unroll 4
  for (int i = 0; i < 64; ++i) mx = fmaxf(mx, sS[row * S_ + j + 16 * i]);
  sAux[row * 16 + j] = mx;         // hq tile no longer needed
  __syncthreads();
  if (j == 0) {
    float mm = -3.4e38f;
    for (int t = 0; t < 16; ++t) mm = fmaxf(mm, sAux[row * 16 + t]);
    sRow[row] = mm;
  }
  __syncthreads();
  float rm = sRow[row];
  float sum = 0.f;
#pragma unroll 4
  for (int i = 0; i < 64; ++i) {
    int idx = row * S_ + j + 16 * i;
    float p = __builtin_amdgcn_exp2f((sS[idx] - rm) * 1.4426950408889634f);
    sS[idx] = p;
    sum += p;
  }
  sAux[row * 16 + j] = sum;
  __syncthreads();
  if (j == 0) {
    float s = 0.f;
    for (int t = 0; t < 16; ++t) s += sAux[row * 16 + t];
    sRow[16 + row] = s;
  }
  __syncthreads();
  float inv = __builtin_amdgcn_rcpf(sRow[16 + row]);
  size_t obase = (qrow + row) * S_;
#pragma unroll 4
  for (int i = 0; i < 64; ++i) {
    int col = j + 16 * i;
    attn[obase + col] = f2bf(sS[row * S_ + col] * inv);
  }
}

// ---------------------------------------------------------------------------
// out = attn @ V : [1024x1024]bf16 x [1024x512]bf16 -> fp32, per batch.
// One block per (b, 16-row q-tile); each wave owns 4 e-tiles (4 independent
// accumulators) and streams the attn A fragments ONCE. Fully unrolled K loop
// with a (4 WMMA | 10 VMEM) steady-state pipeline: the 4 independent WMMAs of
// iteration i overlap all loads of iteration i+1.
// ---------------------------------------------------------------------------
__global__ __launch_bounds__(256) void pv_kernel(
    const u16* __restrict__ attn, const u16* __restrict__ Vt,
    float* __restrict__ out) {
  int lane = threadIdx.x & 31, wave = threadIdx.x >> 5;
  int b  = blockIdx.x >> 6;
  int q0 = (blockIdx.x & 63) << 4;
  size_t arow = (size_t)(b * S_ + q0) * S_;
  int n = lane & 15, hf = lane >> 4;

  size_t vrow[4];
#pragma unroll
  for (int j = 0; j < 4; ++j)
    vrow[j] = (size_t)(b * E_ + (wave + 8 * j) * 16) * S_;

  v8f c0 = {}, c1 = {}, c2 = {}, c3 = {};
  bf16x16 af = load_frag(attn + arow, S_, lane);
#pragma unroll
  for (int kk = 0; kk < 32; ++kk) {
    bf16x16 bf0 = load_frag(Vt + vrow[0] + kk * 32, S_, lane);
    bf16x16 bf1 = load_frag(Vt + vrow[1] + kk * 32, S_, lane);
    bf16x16 bf2 = load_frag(Vt + vrow[2] + kk * 32, S_, lane);
    bf16x16 bf3 = load_frag(Vt + vrow[3] + kk * 32, S_, lane);
    bf16x16 an = af;
    if (kk + 1 < 32) an = load_frag(attn + arow + (kk + 1) * 32, S_, lane);
    c0 = __builtin_amdgcn_wmma_f32_16x16x32_bf16(false, af, false, bf0,
                                                 (short)0, c0, false, false);
    c1 = __builtin_amdgcn_wmma_f32_16x16x32_bf16(false, af, false, bf1,
                                                 (short)0, c1, false, false);
    c2 = __builtin_amdgcn_wmma_f32_16x16x32_bf16(false, af, false, bf2,
                                                 (short)0, c2, false, false);
    c3 = __builtin_amdgcn_wmma_f32_16x16x32_bf16(false, af, false, bf3,
                                                 (short)0, c3, false, false);
    af = an;
  }
  // Pipeline: preload iter0's 10 loads; 30x steady-state (4 WMMA, 10 VMEM);
  // tail iter has 8 loads (no A prefetch); then 4 draining WMMAs.
  // All group sizes are literal constants (builtin requires ICE args).
  SGB(MASK_VMEM_READ, 10, 0);
#pragma unroll
  for (int it = 0; it < 30; ++it) {
    SGB(MASK_WMMA, 4, 0);
    SGB(MASK_VMEM_READ, 10, 0);
  }
  SGB(MASK_WMMA, 4, 0);
  SGB(MASK_VMEM_READ, 8, 0);
  SGB(MASK_WMMA, 4, 0);

#pragma unroll
  for (int j = 0; j < 4; ++j) {
    int e0 = (wave + 8 * j) * 16;
    v8f cc = (j == 0) ? c0 : (j == 1) ? c1 : (j == 2) ? c2 : c3;
#pragma unroll
    for (int r = 0; r < 8; ++r) {
      int m = r + 8 * hf;
      out[(size_t)(b * S_ + q0 + m) * E_ + e0 + n] = cc[r];
    }
  }
}

// ---------------------------------------------------------------------------
// Host-side launch
// ---------------------------------------------------------------------------
extern "C" void kernel_launch(void* const* d_in, const int* in_sizes, int n_in,
                              void* d_out, int out_size, void* d_ws, size_t ws_size,
                              hipStream_t stream) {
  const float* x  = (const float*)d_in[0];
  const float* Wq = (const float*)d_in[1];
  const float* Wk = (const float*)d_in[2];
  const float* Wv = (const float*)d_in[3];
  const float* W1 = (const float*)d_in[4];
  const float* b1 = (const float*)d_in[5];
  const float* W2 = (const float*)d_in[6];
  const float* b2 = (const float*)d_in[7];
  float* out = (float*)d_out;

  const int BSE = B_ * S_ * E_;   // 2,097,152
  const int EE  = E_ * E_;        // 262,144
  const int BSH = B_ * S_ * H_;   // 65,536
  const int BSS = B_ * S_ * S_;   // 4,194,304

  char* ws = (char*)d_ws;
  size_t off = 0;
  auto alloc = [&](size_t bytes) -> void* {
    void* p = ws + off;
    off += (bytes + 255) & ~(size_t)255;
    return p;
  };
  u16*   xbf   = (u16*)alloc((size_t)BSE * 2);
  u16*   Wqb   = (u16*)alloc((size_t)EE * 2);
  u16*   Wkb   = (u16*)alloc((size_t)EE * 2);
  u16*   Wvb   = (u16*)alloc((size_t)EE * 2);
  u16*   W1qb  = (u16*)alloc((size_t)H_ * E_ * 2);
  u16*   W1kb  = (u16*)alloc((size_t)H_ * E_ * 2);
  u16*   Qb    = (u16*)alloc((size_t)BSE * 2);
  u16*   Kb    = (u16*)alloc((size_t)BSE * 2);
  u16*   Vt    = (u16*)alloc((size_t)BSE * 2);
  float* hqf   = (float*)alloc((size_t)BSH * 4);
  float* hkf   = (float*)alloc((size_t)BSH * 4);
  u16*   attnb = (u16*)alloc((size_t)BSS * 2);

  // 1) bf16 conversions
  cvt_bf16<<<(BSE + 255) / 256, 256, 0, stream>>>(x,  xbf, BSE);
  cvt_bf16<<<(EE  + 255) / 256, 256, 0, stream>>>(Wq, Wqb, EE);
  cvt_bf16<<<(EE  + 255) / 256, 256, 0, stream>>>(Wk, Wkb, EE);
  cvt_bf16<<<(EE  + 255) / 256, 256, 0, stream>>>(Wv, Wvb, EE);
  split_w1<<<(H_ * E_ + 255) / 256, 256, 0, stream>>>(W1, W1qb, W1kb);

  // 2) Q,K (row-major bf16) and V (transposed bf16) projections
  qkv_kernel<<<B_ * (S_ / 16), 256, 0, stream>>>(xbf, Wqb, Wkb, Wvb, Qb, Kb, Vt);

  // 3) hq = Q@W1q^T, hk = K@W1k^T (fp32)
  hqk_kernel<<<(B_ * S_ / 16) / 8, 256, 0, stream>>>(Qb, W1qb, hqf);
  hqk_kernel<<<(B_ * S_ / 16) / 8, 256, 0, stream>>>(Kb, W1kb, hkf);

  // 4) fused scores + tanh-MLP est + row softmax -> attn bf16
  attn_scores_kernel<<<B_ * (S_ / 16), 256, 0, stream>>>(
      Qb, Kb, hqf, hkf, b1, W2, b2, attnb);

  // 5) out = attn @ V
  pv_kernel<<<B_ * (S_ / 16), 256, 0, stream>>>(attnb, Vt, out);

  (void)in_sizes; (void)n_in; (void)out_size; (void)ws_size;
}
